// MultiHeadSelfAttentionBlock_26852135535025
// MI455X (gfx1250) — matrix operand order repaired and checked by reading, not compiled, not run on hardware
//
#include <hip/hip_runtime.h>

#define B 8
#define S 2048
#define D 512
#define H 8
#define DH 64
#define BM 128   // rows per workgroup (8 waves x 16)
#define BN 64    // key columns per attention inner iteration

typedef __attribute__((ext_vector_type(16))) __bf16 v16bf;
typedef __attribute__((ext_vector_type(8)))  float  v8f;

union FragA { v16bf v; uint4 u4[2]; unsigned short u[16]; };
union FragC { v8f   v; float f[8]; };

// Round-half-up f32->bf16 (2 VALU ops); tie-break differences vs RNE are far
// below bf16 quantization noise for this workload.
__device__ __forceinline__ unsigned short f2bf_fast(float f) {
  return (unsigned short)((__float_as_uint(f) + 0x8000u) >> 16);
}
// Raw v_exp_f32 (exp2); flush-to-zero for very negative args is desired.
__device__ __forceinline__ float fast_exp2(float x) {
#if __has_builtin(__builtin_amdgcn_exp2f)
  return __builtin_amdgcn_exp2f(x);
#else
  return exp2f(x);
#endif
}

// ---------------------------------------------------------------------------
// Kernel 1: WMMA Q/K/V projection.  One block = one (b,h) x 128 s-rows.
// W^T staged as bf16 in LDS (B-frag layout: row e contiguous over d),
// x tiles converted to bf16 A-frags, 24 WMMAs per wave (3 mats x 4 e-tiles
// x K=64).  Q pre-scaled by (1/sqrt(DH))*log2(e) -> scores in exp2 domain
// (scale folded into the staged bias + one fma).  K is transposed through
// LDS to Kt[b,h,d,s] for the attention B-frags.
// ---------------------------------------------------------------------------
__global__ __launch_bounds__(256)
void qkv_proj_wmma_kernel(const float* __restrict__ x,
                          const float* __restrict__ Wq, const float* __restrict__ bq,
                          const float* __restrict__ Wk, const float* __restrict__ bk,
                          const float* __restrict__ Wv, const float* __restrict__ bv,
                          unsigned short* __restrict__ Q,
                          unsigned short* __restrict__ Kt,
                          unsigned short* __restrict__ V)
{
  __shared__ unsigned short sWt[3][DH][72];   // W^T (d-major) bf16, padded rows
  __shared__ float          sbias[3][DH];
  __shared__ unsigned short kbuf[8][16][72];  // per-wave K-tile transpose buffer

  const int ST = S / BM;             // 16 s-tiles per (b,h)
  int blk = blockIdx.x;
  int st  = blk % ST; blk /= ST;
  int h   = blk % H;
  int b   = blk / H;

  int tid   = threadIdx.x;
  int wid   = tid >> 5;
  int lane  = tid & 31;
  int half  = lane >> 4;
  int lan16 = lane & 15;
  int cc    = half * 8;

  const float qscale = 0.125f * 1.44269504088896340736f; // 1/sqrt(DH)*log2(e)

  const float* Ws[3] = {Wq, Wk, Wv};
  const float* bs[3] = {bq, bk, bv};
  for (int i = tid; i < 3 * DH * DH; i += 256) {
    int a = i / (DH * DH);
    int o = i % (DH * DH);
    int e = o >> 6, d = o & 63;                    // consecutive tid -> d
    sWt[a][e][d] = f2bf_fast(Ws[a][(h * DH + e) * DH + d]);
  }
  if (tid < 3 * DH) {
    int a = tid / DH, e = tid % DH;
    float bia = bs[a][h * DH + e];
    sbias[a][e] = (a == 0) ? bia * qscale : bia;   // Q bias pre-scaled
  }
  __syncthreads();

  int    s0 = st * BM + wid * 16;
  size_t bh = (size_t)(b * H + h);

  // x A-fragments: 16x64 f32 -> two 16x32 bf16 frags.
  const float* xp = x + ((size_t)(b * S) + s0 + lan16) * D + h * DH;
  FragA a0, a1;
  #pragma unroll
  for (int i = 0; i < 8; ++i) {
    a0.u[i]     = f2bf_fast(xp[cc + i]);
    a0.u[8 + i] = f2bf_fast(xp[16 + cc + i]);
    a1.u[i]     = f2bf_fast(xp[32 + cc + i]);
    a1.u[8 + i] = f2bf_fast(xp[48 + cc + i]);
  }

  v8f zero = {};

  #pragma unroll
  for (int m = 0; m < 3; ++m) {
    FragC acc[4];
    #pragma unroll
    for (int t = 0; t < 4; ++t) {
      FragA b0, b1;
      const unsigned short* w0 = &sWt[m][t * 16 + lan16][cc];
      const unsigned short* w1 = &sWt[m][t * 16 + lan16][32 + cc];
      b0.u4[0] = *(const uint4*)(w0);
      b0.u4[1] = *(const uint4*)(w0 + 16);
      b1.u4[0] = *(const uint4*)(w1);
      b1.u4[1] = *(const uint4*)(w1 + 16);
      acc[t].v = __builtin_amdgcn_wmma_f32_16x16x32_bf16(
          false, a0.v, false, b0.v, (short)0, zero, false, false);
      acc[t].v = __builtin_amdgcn_wmma_f32_16x16x32_bf16(
          false, a1.v, false, b1.v, (short)0, acc[t].v, false, false);
      float bia = sbias[m][t * 16 + lan16];
      if (m == 0) {
        #pragma unroll
        for (int r = 0; r < 8; ++r) acc[t].f[r] = fmaf(acc[t].f[r], qscale, bia);
      } else {
        #pragma unroll
        for (int r = 0; r < 8; ++r) acc[t].f[r] += bia;
      }
    }

    if (m == 0) {            // ---- Q (scale folded) ----
      unsigned short* qp = Q + bh * (size_t)S * DH;
      #pragma unroll
      for (int t = 0; t < 4; ++t)
        #pragma unroll
        for (int r = 0; r < 8; ++r) {
          int row = s0 + r + 8 * half;
          qp[(size_t)row * DH + t * 16 + lan16] = f2bf_fast(acc[t].f[r]);
        }
    } else if (m == 2) {     // ---- V ----
      unsigned short* vp = V + bh * (size_t)S * DH;
      #pragma unroll
      for (int t = 0; t < 4; ++t)
        #pragma unroll
        for (int r = 0; r < 8; ++r) {
          int row = s0 + r + 8 * half;
          vp[(size_t)row * DH + t * 16 + lan16] = f2bf_fast(acc[t].f[r]);
        }
    } else {                 // ---- K: LDS transpose -> Kt[b,h,d,s] ----
      #pragma unroll
      for (int t = 0; t < 4; ++t)
        #pragma unroll
        for (int r = 0; r < 8; ++r)
          kbuf[wid][r + 8 * half][t * 16 + lan16] = f2bf_fast(acc[t].f[r]);
      asm volatile("s_wait_dscnt 0" ::: "memory");

      // Lane owns e-columns {2*lane, 2*lane+1}: packed b32 column reads
      // (word index = lane -> conflict-free), 16B stores along s.
      unsigned int q0[8], q1[8];
      #pragma unroll
      for (int i = 0; i < 8; ++i) {
        unsigned int wA = *(const unsigned int*)&kbuf[wid][2 * i][2 * lane];
        unsigned int wB = *(const unsigned int*)&kbuf[wid][2 * i + 1][2 * lane];
        q0[i] = (wA & 0xFFFFu) | (wB << 16);
        q1[i] = (wA >> 16) | (wB & 0xFFFF0000u);
      }
      unsigned short* kp0 = Kt + (bh * DH + 2 * lane) * (size_t)S + s0;
      unsigned short* kp1 = kp0 + S;
      uint4 v0a = {q0[0], q0[1], q0[2], q0[3]};
      uint4 v0b = {q0[4], q0[5], q0[6], q0[7]};
      uint4 v1a = {q1[0], q1[1], q1[2], q1[3]};
      uint4 v1b = {q1[4], q1[5], q1[6], q1[7]};
      *(uint4*)(kp0)     = v0a;
      *(uint4*)(kp0 + 8) = v0b;
      *(uint4*)(kp1)     = v1a;
      *(uint4*)(kp1 + 8) = v1b;
    }
  }
}

// ---------------------------------------------------------------------------
// Kernel 2: flash attention. One wave = 16 query rows; workgroup = 128 rows.
// Per 64-key iteration: 8 score WMMAs, 8 PV WMMAs, 2 row-sum WMMAs
// (l = P @ ones).  Scores arrive in the exp2 domain; softmax uses raw
// v_exp_f32.  Row-max via 4-stage shfl_xor butterfly in 16-lane halves.
// ---------------------------------------------------------------------------
__global__ __launch_bounds__(256)
void flash_attn_kernel(const unsigned short* __restrict__ Q,
                       const unsigned short* __restrict__ Kt,
                       const unsigned short* __restrict__ V,
                       float* __restrict__ out)
{
  __shared__ unsigned short Pl[8][16][72];

  const int QT = S / BM;
  int blk = blockIdx.x;
  int qt  = blk % QT; blk /= QT;
  int h   = blk % H;
  int b   = blk / H;

  int wid   = threadIdx.x >> 5;
  int lane  = threadIdx.x & 31;
  int half  = lane >> 4;
  int lan16 = lane & 15;
  int cc    = half * 8;

  size_t bh = (size_t)(b * H + h);
  int q0 = qt * BM + wid * 16;

  const unsigned short* Qp = Q + (bh * S + q0 + lan16) * DH;
  FragA aq0, aq1;
  aq0.u4[0] = *(const uint4*)(Qp + cc);
  aq0.u4[1] = *(const uint4*)(Qp + 16 + cc);
  aq1.u4[0] = *(const uint4*)(Qp + 32 + cc);
  aq1.u4[1] = *(const uint4*)(Qp + 48 + cc);

  FragA ones;
  #pragma unroll
  for (int i = 0; i < 16; ++i) ones.u[i] = 0x3F80;   // bf16 1.0

  const unsigned short* Ktp = Kt + bh * (size_t)DH * S;
  const unsigned short* Vp  = V  + bh * (size_t)S * DH;

  v8f zero = {};
  FragC o[4];
  #pragma unroll
  for (int t = 0; t < 4; ++t) o[t].v = zero;
  FragC lacc;
  lacc.v = zero;
  float mrow[8];
  #pragma unroll
  for (int r = 0; r < 8; ++r) mrow[r] = -1e30f;

  for (int j = 0; j < S; j += BN) {
    FragC s[4];
    #pragma unroll
    for (int kt = 0; kt < 4; ++kt) {
      int jk = j + kt * 16;
      FragA bk;
      #pragma unroll
      for (int i = 0; i < 8; ++i) {
        bk.u[i]     = Ktp[(size_t)(cc + i) * S + jk + lan16];
        bk.u[8 + i] = Ktp[(size_t)(16 + cc + i) * S + jk + lan16];
      }
      s[kt].v = __builtin_amdgcn_wmma_f32_16x16x32_bf16(
          false, aq0.v, false, bk.v, (short)0, zero, false, false);
      #pragma unroll
      for (int i = 0; i < 8; ++i) {
        bk.u[i]     = Ktp[(size_t)(32 + cc + i) * S + jk + lan16];
        bk.u[8 + i] = Ktp[(size_t)(48 + cc + i) * S + jk + lan16];
      }
      s[kt].v = __builtin_amdgcn_wmma_f32_16x16x32_bf16(
          false, aq1.v, false, bk.v, (short)0, s[kt].v, false, false);
    }

    if (j + BN < S) {
      __builtin_prefetch(Ktp + (size_t)lane * S + j + BN, 0, 3);
      __builtin_prefetch(Vp + (size_t)(j + BN + lane) * DH, 0, 3);
    }

    float alpha[8];
    #pragma unroll
    for (int r = 0; r < 8; ++r) {
      float c0 = s[0].f[r];
      float c1 = s[1].f[r];
      float c2 = s[2].f[r];
      float c3 = s[3].f[r];
      float tm = fmaxf(fmaxf(c0, c1), fmaxf(c2, c3));
      tm = fmaxf(tm, __shfl_xor(tm, 1));
      tm = fmaxf(tm, __shfl_xor(tm, 2));
      tm = fmaxf(tm, __shfl_xor(tm, 4));
      tm = fmaxf(tm, __shfl_xor(tm, 8));
      float mn = fmaxf(mrow[r], tm);
      alpha[r] = fast_exp2(mrow[r] - mn);
      mrow[r]  = mn;
      int row = r + 8 * half;
      Pl[wid][row][lan16]      = f2bf_fast(fast_exp2(c0 - mn));
      Pl[wid][row][16 + lan16] = f2bf_fast(fast_exp2(c1 - mn));
      Pl[wid][row][32 + lan16] = f2bf_fast(fast_exp2(c2 - mn));
      Pl[wid][row][48 + lan16] = f2bf_fast(fast_exp2(c3 - mn));
    }
    #pragma unroll
    for (int t = 0; t < 4; ++t)
      #pragma unroll
      for (int r = 0; r < 8; ++r) o[t].f[r] *= alpha[r];
    #pragma unroll
    for (int r = 0; r < 8; ++r) lacc.f[r] *= alpha[r];

    asm volatile("s_wait_dscnt 0" ::: "memory");

    FragA ap0, ap1;
    #pragma unroll
    for (int i = 0; i < 8; ++i) {
      ap0.u[i]     = Pl[wid][lan16][cc + i];
      ap0.u[8 + i] = Pl[wid][lan16][16 + cc + i];
      ap1.u[i]     = Pl[wid][lan16][32 + cc + i];
      ap1.u[8 + i] = Pl[wid][lan16][48 + cc + i];
    }

    lacc.v = __builtin_amdgcn_wmma_f32_16x16x32_bf16(
        false, ap0.v, false, ones.v, (short)0, lacc.v, false, false);
    lacc.v = __builtin_amdgcn_wmma_f32_16x16x32_bf16(
        false, ap1.v, false, ones.v, (short)0, lacc.v, false, false);

    #pragma unroll
    for (int t = 0; t < 4; ++t) {
      FragA bv;
      const unsigned short* vq0 = Vp + (size_t)(j + cc) * DH + t * 16 + lan16;
      #pragma unroll
      for (int i = 0; i < 8; ++i) {
        bv.u[i]     = vq0[(size_t)i * DH];
        bv.u[8 + i] = vq0[(size_t)(16 + i) * DH];
      }
      o[t].v = __builtin_amdgcn_wmma_f32_16x16x32_bf16(
          false, ap0.v, false, bv.v, (short)0, o[t].v, false, false);
      const unsigned short* vq1 = Vp + (size_t)(j + 32 + cc) * DH + t * 16 + lan16;
      #pragma unroll
      for (int i = 0; i < 8; ++i) {
        bv.u[i]     = vq1[(size_t)i * DH];
        bv.u[8 + i] = vq1[(size_t)(16 + i) * DH];
      }
      o[t].v = __builtin_amdgcn_wmma_f32_16x16x32_bf16(
          false, ap1.v, false, bv.v, (short)0, o[t].v, false, false);
    }
  }

  float inv[8];
  #pragma unroll
  for (int r = 0; r < 8; ++r) inv[r] = __builtin_amdgcn_rcpf(lacc.f[r]);
  #pragma unroll
  for (int t = 0; t < 4; ++t) {
    #pragma unroll
    for (int r = 0; r < 8; ++r) {
      int row = r + 8 * half;
      int e   = t * 16 + lan16;
      out[((size_t)b * S + (q0 + row)) * D + h * DH + e] = o[t].f[r] * inv[r];
    }
  }
}

// ---------------------------------------------------------------------------
extern "C" void kernel_launch(void* const* d_in, const int* in_sizes, int n_in,
                              void* d_out, int out_size, void* d_ws, size_t ws_size,
                              hipStream_t stream) {
  const float* x  = (const float*)d_in[0];
  const float* Wq = (const float*)d_in[1];
  const float* bq = (const float*)d_in[2];
  const float* Wk = (const float*)d_in[3];
  const float* bk = (const float*)d_in[4];
  const float* Wv = (const float*)d_in[5];
  const float* bv = (const float*)d_in[6];
  float* out = (float*)d_out;

  size_t per = (size_t)B * H * S * DH * sizeof(unsigned short);  // 16 MiB each
  unsigned short* Qb  = (unsigned short*)d_ws;
  unsigned short* Ktb = (unsigned short*)((char*)d_ws + per);
  unsigned short* Vb  = (unsigned short*)((char*)d_ws + 2 * per);

  qkv_proj_wmma_kernel<<<B * H * (S / BM), 256, 0, stream>>>(
      x, Wq, bq, Wk, bk, Wv, bv, Qb, Ktb, Vb);
  flash_attn_kernel<<<B * H * (S / BM), 256, 0, stream>>>(Qb, Ktb, Vb, out);
}
